// WindowAttention_62380105007725
// MI455X (gfx1250) — compile-verified, matrix operand order checked by reading, not run on hardware
//
#include <hip/hip_runtime.h>

// ---------------------------------------------------------------------------
// Swin window attention, fused per-window, bf16 WMMA (fp32 accumulate),
// TDM (tensor_load_to_lds) staging of the x window.
// B_=4096 windows, N=49 (pad 64), C=256, H=8 heads, hd=32, nW=64 masks.
// ---------------------------------------------------------------------------

typedef __attribute__((ext_vector_type(16))) __bf16    v16bf;
typedef __attribute__((ext_vector_type(8)))  float     v8f;
typedef __attribute__((ext_vector_type(4)))  unsigned  v4u;
typedef __attribute__((ext_vector_type(8)))  int       v8i;
typedef __attribute__((ext_vector_type(4)))  int       v4i;

#define N_TOK   49
#define N_PAD   64
#define CDIM    256
#define HEADS   8
#define HDIM    32
#define QKV_OUT 768
#define WINB    4096
#define SCALE_Q 0.17677669529663687f  // 32^-0.5

// LDS carve (bytes):
//   [0, 65536)         : x window bf16 (64x256=32KB) UNION P probs (8x64x64 bf16)
//   [65536, 163840)    : Q|K|V contiguous [24][64][32] bf16 (Q reused for O)
//   [163840, 215040)   : fp32 staging for TDM x-window load (49x256x4 = 50176B)
#define QKV_OFF    65536
#define STAGE_OFF  163840
#define SMEM_BYTES 215040

#if __has_builtin(__builtin_amdgcn_tensor_load_to_lds) && \
    __has_builtin(__builtin_amdgcn_s_wait_tensorcnt)
#define HAVE_TDM 1
#else
#define HAVE_TDM 0
#endif

static __device__ __forceinline__ __bf16 f2bf(float f) {
  unsigned u = __builtin_bit_cast(unsigned, f);
  unsigned r = u + 0x7FFFu + ((u >> 16) & 1u);            // round-to-nearest-even
  return __builtin_bit_cast(__bf16, (unsigned short)(r >> 16));
}

// K index of fragment register r (0..7), element 0; element 1 adds +1.
// Matches ISA 16-bit A 16x32 layout; lane-group (+8) folded into pointer.
static __device__ __forceinline__ int kmap(int r) {
  return ((r & 4) << 2) | ((r & 3) << 1);
}

// Load a 16x32 bf16 A-fragment: element (row, kBase+k).
static __device__ __forceinline__ v16bf loadA(const __bf16* base, int row, int ld,
                                              int kBase, int g) {
  v16bf a;
  const __bf16* p = base + row * ld + kBase + (g << 3);
  #pragma unroll
  for (int r = 0; r < 8; ++r) {
    int k0 = kmap(r);
    a[2 * r]     = p[k0];
    a[2 * r + 1] = p[k0 + 1];
  }
  return a;
}

static __device__ __forceinline__ v8f wmma_bf16(v16bf a, v16bf b, v8f c) {
  return __builtin_amdgcn_wmma_f32_16x16x32_bf16(false, a, false, b, (short)0, c,
                                                 false, false);
}

// ---------------------------------------------------------------------------
// Prologue: fp32 -> bf16 weight conversion into d_ws (L2-resident afterwards).
// ---------------------------------------------------------------------------
__global__ void cvt_weights_kernel(const float* __restrict__ qkvw,
                                   const float* __restrict__ projw,
                                   __bf16* __restrict__ wq,
                                   __bf16* __restrict__ wp) {
  int i = blockIdx.x * blockDim.x + threadIdx.x;
  if (i < QKV_OUT * CDIM) wq[i] = f2bf(qkvw[i]);
  if (i < CDIM * CDIM)    wp[i] = f2bf(projw[i]);
}

// ---------------------------------------------------------------------------
// Fused per-window kernel: 4096 blocks x 256 threads (8 wave32).
// ---------------------------------------------------------------------------
__global__ __launch_bounds__(256, 1)
void swin_window_attn_kernel(const float* __restrict__ x,
                             const float* __restrict__ mask,
                             const float* __restrict__ qkv_b,
                             const float* __restrict__ proj_b,
                             const float* __restrict__ rpb,
                             const __bf16* __restrict__ wq,
                             const __bf16* __restrict__ wp,
                             float* __restrict__ out) {
  extern __shared__ char smem[];
  __bf16* xsP  = (__bf16*)smem;                       // [64][256], phase 1-2
  __bf16* pP   = (__bf16*)smem;                       // [8][64][64], phase 3
  __bf16* qArr = (__bf16*)(smem + QKV_OFF);           // [8][64][32] (O after attn)
  __bf16* kArr = qArr + HEADS * N_PAD * HDIM;
  __bf16* vArr = kArr + HEADS * N_PAD * HDIM;

  const int tid  = threadIdx.x;
  const int w    = tid >> 5;         // wave id 0..7
  const int l16  = tid & 15;         // lane within 16-lane half
  const int g    = (tid >> 4) & 1;   // lane-group (0: lanes 0-15, 1: 16-31)
  const int b    = blockIdx.x;       // window id

  // ---------------- Phase 1: stage x window (zero-padded) as bf16 -----------
#if HAVE_TDM
  {
    // Wave 0 issues one TDM descriptor: 2D tile 256x49 fp32, row stride 256,
    // HBM -> LDS staging. Completion tracked on the issuing wave's TENSORcnt;
    // other waves synchronize at the workgroup barrier.
    if (w == 0) {
      const unsigned long long ga =
          (unsigned long long)(uintptr_t)(x + (size_t)b * N_TOK * CDIM);
      const unsigned ldsAddr = (unsigned)(uintptr_t)(smem + STAGE_OFF);

      v4u g0;
      g0[0] = 1u;                                   // count=1, user load
      g0[1] = ldsAddr;                              // lds_addr
      g0[2] = (unsigned)(ga & 0xFFFFFFFFu);         // global_addr[31:0]
      g0[3] = (unsigned)((ga >> 32) & 0x01FFFFFFu)  // global_addr[56:32]
              | (2u << 30);                         // type = 2 ("image")

      v8i g1;
      g1[0] = (int)(2u << 16);                      // data_size = 4 bytes
      g1[1] = (int)((unsigned)CDIM << 16);          // tensor_dim0 = 256
      g1[2] = (int)((unsigned)N_TOK << 16);         // tensor_dim1 = 49
      g1[3] = (int)((unsigned)CDIM << 16);          // tile_dim0 = 256
      g1[4] = (int)N_TOK;                           // tile_dim1 = 49, tile_dim2 = 0
      g1[5] = (int)CDIM;                            // tensor_dim0_stride = 256
      g1[6] = 0;
      g1[7] = 0;

      v4i g2 = {0, 0, 0, 0};
      v4i g3 = {0, 0, 0, 0};
#if __clang_major__ >= 23
      v8i g4 = {0, 0, 0, 0, 0, 0, 0, 0};
      __builtin_amdgcn_tensor_load_to_lds(g0, g1, g2, g3, g4, 0);
#else
      __builtin_amdgcn_tensor_load_to_lds(g0, g1, g2, g3, 0);
#endif
      __builtin_amdgcn_s_wait_tensorcnt(0);
    }
    __syncthreads();
    const float* stage = (const float*)(smem + STAGE_OFF);
    for (int idx = tid; idx < N_PAD * CDIM; idx += 256) {
      int row = idx >> 8;
      float f = (row < N_TOK) ? stage[idx] : 0.0f;
      xsP[idx] = f2bf(f);
    }
  }
#else
  {
    const float* xb = x + (size_t)b * N_TOK * CDIM;
    for (int idx = tid; idx < N_PAD * CDIM; idx += 256) {
      int row = idx >> 8;
      float f = (row < N_TOK) ? xb[idx] : 0.0f;
      xsP[idx] = f2bf(f);
    }
  }
#endif
  __syncthreads();

  // ---------------- Phase 2: QKV GEMM 64x768, K=256 -------------------------
  // 48 column tiles of 16; wave w handles jt = w, w+8, ..., w+40.
  for (int t = 0; t < 6; ++t) {
    const int jt    = w + 8 * t;
    const int jBase = jt * 16;
    v8f acc[4];
    #pragma unroll
    for (int mi = 0; mi < 4; ++mi)
      acc[mi] = (v8f){0.f,0.f,0.f,0.f,0.f,0.f,0.f,0.f};

    for (int kc = 0; kc < 8; ++kc) {
      const int kBase = kc * 32;
      // B fragment: B[k][n] = W[(jBase+n)*256 + kBase+k]  (W row-major (768,256))
      v16bf bf_;
      {
        const __bf16* p = wq + (size_t)(jBase + l16) * CDIM + kBase + (g << 3);
        #pragma unroll
        for (int r = 0; r < 8; ++r) {
          int k0 = kmap(r);
          bf_[2 * r]     = p[k0];
          bf_[2 * r + 1] = p[k0 + 1];
        }
      }
      #pragma unroll
      for (int mi = 0; mi < 4; ++mi) {
        v16bf af = loadA(xsP, mi * 16 + l16, CDIM, kBase, g);
        acc[mi] = wmma_bf16(af, bf_, acc[mi]);
      }
    }
    // Branchless epilogue: Q|K|V are contiguous in LDS, so the destination is
    // qArr[(j>>5)*2048 + row*32 + (j&31)] for all j in 0..767.
    const int   j   = jBase + l16;
    const float bj  = qkv_b[j];
    const float scl = (j < 256) ? SCALE_Q : 1.0f;
    __bf16* dst = qArr + (j >> 5) * (N_PAD * HDIM) + (j & 31) + (g << 3) * HDIM;
    #pragma unroll
    for (int mi = 0; mi < 4; ++mi) {
      #pragma unroll
      for (int v = 0; v < 8; ++v) {
        float val = (acc[mi][v] + bj) * scl;
        dst[(mi * 16 + v) * HDIM] = f2bf(val);
      }
    }
  }
  __syncthreads();

  // ---------------- Phase 3: attention, one wave per head -------------------
  {
    const int h = w;
    const __bf16* qH = qArr + h * (N_PAD * HDIM);
    const __bf16* kH = kArr + h * (N_PAD * HDIM);
    const __bf16* vH = vArr + h * (N_PAD * HDIM);
    __bf16*       pH = pP   + h * (N_PAD * N_PAD);
    const float*  mW = mask + (size_t)(b & 63) * N_TOK * N_TOK;

    // S = Q(64x32) @ K^T(32x64): 4x4 tiles, single K-chunk of 32.
    v8f S[4][4];
    #pragma unroll
    for (int mi = 0; mi < 4; ++mi) {
      v16bf aq = loadA(qH, mi * 16 + l16, HDIM, 0, g);
      #pragma unroll
      for (int nj = 0; nj < 4; ++nj) {
        v16bf bk;  // B[d][n] = K[(nj*16+n)][d]
        {
          const __bf16* p = kH + (size_t)(nj * 16 + l16) * HDIM + (g << 3);
          #pragma unroll
          for (int r = 0; r < 8; ++r) {
            int k0 = kmap(r);
            bk[2 * r]     = p[k0];
            bk[2 * r + 1] = p[k0 + 1];
          }
        }
        v8f z = (v8f){0.f,0.f,0.f,0.f,0.f,0.f,0.f,0.f};
        S[mi][nj] = wmma_bf16(aq, bk, z);
      }
    }

    // Bias + mask + row softmax (rows live across 16-lane halves -> shfl_xor).
    #pragma unroll
    for (int mi = 0; mi < 4; ++mi) {
      #pragma unroll
      for (int v = 0; v < 8; ++v) {
        const int i = mi * 16 + v + 8 * g;
        float xv[4];
        #pragma unroll
        for (int nj = 0; nj < 4; ++nj) {
          const int j = nj * 16 + l16;
          float s = S[mi][nj][v];
          if (i < N_TOK && j < N_TOK) {
            int ri = i / 7, ci = i - ri * 7;
            int rj = j / 7, cj = j - rj * 7;
            int rel = (ri - rj + 6) * 13 + (ci - cj + 6);
            s += rpb[rel * HEADS + h] + mW[i * N_TOK + j];
          }
          if (j >= N_TOK) s = -1e30f;
          xv[nj] = s;
        }
        float m = fmaxf(fmaxf(xv[0], xv[1]), fmaxf(xv[2], xv[3]));
        #pragma unroll
        for (int off = 1; off < 16; off <<= 1)
          m = fmaxf(m, __shfl_xor(m, off, 32));
        float e[4], sum = 0.f;
        #pragma unroll
        for (int nj = 0; nj < 4; ++nj) { e[nj] = __expf(xv[nj] - m); sum += e[nj]; }
        #pragma unroll
        for (int off = 1; off < 16; off <<= 1)
          sum += __shfl_xor(sum, off, 32);
        const float inv = 1.0f / sum;
        #pragma unroll
        for (int nj = 0; nj < 4; ++nj)
          pH[i * N_PAD + nj * 16 + l16] = f2bf(e[nj] * inv);
      }
    }

    // O = P(64x64) @ V(64x32); O overwrites the Q region (same-wave LDS is
    // in-order, and only this wave touches head h).
    #pragma unroll
    for (int mi = 0; mi < 4; ++mi) {
      v16bf pa0 = loadA(pH, mi * 16 + l16, N_PAD, 0, g);
      v16bf pa1 = loadA(pH, mi * 16 + l16, N_PAD, 32, g);
      #pragma unroll
      for (int ni = 0; ni < 2; ++ni) {
        v8f o = (v8f){0.f,0.f,0.f,0.f,0.f,0.f,0.f,0.f};
        #pragma unroll
        for (int kc = 0; kc < 2; ++kc) {
          v16bf bv;  // B[j][d] = V[kc*32 + j][ni*16 + d]
          {
            const __bf16* p = vH + (size_t)(kc * 32 + (g << 3)) * HDIM + ni * 16 + l16;
            #pragma unroll
            for (int r = 0; r < 8; ++r) {
              int k0 = kmap(r);
              bv[2 * r]     = p[(size_t)k0 * HDIM];
              bv[2 * r + 1] = p[(size_t)(k0 + 1) * HDIM];
            }
          }
          o = wmma_bf16(kc == 0 ? pa0 : pa1, bv, o);
        }
        __bf16* oH = qArr + h * (N_PAD * HDIM);
        #pragma unroll
        for (int v = 0; v < 8; ++v) {
          int row = mi * 16 + v + 8 * g;
          oH[row * HDIM + ni * 16 + l16] = f2bf(o[v]);
        }
      }
    }
  }
  __syncthreads();

  // ---------------- Phase 4: proj GEMM 64x256, K=256, + bias ----------------
  // O lives in qArr as [head(=k-chunk)][row][32]; global col c = kc*32 + kk.
  for (int t = 0; t < 2; ++t) {
    const int jt    = w * 2 + t;      // 0..15 column tiles
    const int jBase = jt * 16;
    v8f acc[4];
    #pragma unroll
    for (int mi = 0; mi < 4; ++mi)
      acc[mi] = (v8f){0.f,0.f,0.f,0.f,0.f,0.f,0.f,0.f};

    for (int kc = 0; kc < 8; ++kc) {
      v16bf bw;  // B[k][n] = projW[(jBase+n)*256 + kc*32 + k]
      {
        const __bf16* p = wp + (size_t)(jBase + l16) * CDIM + kc * 32 + (g << 3);
        #pragma unroll
        for (int r = 0; r < 8; ++r) {
          int k0 = kmap(r);
          bw[2 * r]     = p[k0];
          bw[2 * r + 1] = p[k0 + 1];
        }
      }
      const __bf16* oC = qArr + kc * (N_PAD * HDIM);
      #pragma unroll
      for (int mi = 0; mi < 4; ++mi) {
        v16bf af = loadA(oC, mi * 16 + l16, HDIM, 0, g);
        acc[mi] = wmma_bf16(af, bw, acc[mi]);
      }
    }
    const float pb = proj_b[jBase + l16];
    #pragma unroll
    for (int mi = 0; mi < 4; ++mi) {
      #pragma unroll
      for (int v = 0; v < 8; ++v) {
        int i = mi * 16 + v + 8 * g;
        if (i < N_TOK)
          out[((size_t)b * N_TOK + i) * CDIM + jBase + l16] = acc[mi][v] + pb;
      }
    }
  }
}

// ---------------------------------------------------------------------------
extern "C" void kernel_launch(void* const* d_in, const int* in_sizes, int n_in,
                              void* d_out, int out_size, void* d_ws, size_t ws_size,
                              hipStream_t stream) {
  const float* x     = (const float*)d_in[0];
  const float* mask  = (const float*)d_in[1];
  const float* qkvw  = (const float*)d_in[2];
  const float* qkvb  = (const float*)d_in[3];
  const float* projw = (const float*)d_in[4];
  const float* projb = (const float*)d_in[5];
  const float* rpb   = (const float*)d_in[6];
  float*       out   = (float*)d_out;

  __bf16* wq = (__bf16*)d_ws;                                       // 768*256 bf16
  __bf16* wp = (__bf16*)((char*)d_ws + (size_t)QKV_OUT * CDIM * 2); // 256*256 bf16

  cvt_weights_kernel<<<(QKV_OUT * CDIM + 255) / 256, 256, 0, stream>>>(qkvw, projw,
                                                                       wq, wp);

  (void)hipFuncSetAttribute((const void*)swin_window_attn_kernel,
                            hipFuncAttributeMaxDynamicSharedMemorySize, SMEM_BYTES);

  swin_window_attn_kernel<<<WINB, 256, SMEM_BYTES, stream>>>(x, mask, qkvb, projb,
                                                             rpb, wq, wp, out);
}
